// SeqMoELogits_17265768529997
// MI455X (gfx1250) — matrix operand order repaired and checked by reading, not compiled, not run on hardware
//
#include <hip/hip_runtime.h>
#include <hip/hip_bf16.h>
#include <math.h>

#define Bt 2048
#define Dd 1024
#define Ee 8
#define Hh 4096
#define Cc 1024
#define HCHUNK 128

#define KT1 (Dd / 32)   // 32  k-tiles in GEMM1
#define NT1 (Hh / 16)   // 256 n-tiles in GEMM1
#define KT2 (Hh / 32)   // 128 k-tiles in GEMM2
#define NT2 (Cc / 16)   // 64  n-tiles in GEMM2

typedef __attribute__((ext_vector_type(16))) __bf16 v16bf;
typedef __attribute__((ext_vector_type(8)))  float  v8f;

__device__ __forceinline__ v8f wmma_bf16(v16bf a, v16bf b, v8f c) {
    return __builtin_amdgcn_wmma_f32_16x16x32_bf16(
        /*neg_a=*/false, a, /*neg_b=*/false, b,
        /*c_mod=*/(short)0, c, /*reuse_a=*/false, /*reuse_b=*/false);
}

// A fragment (16x32 bf16, M = lane&15, koff = (lane>>4)*16):
// per-lane 16 values = contiguous run [p, p+16) of the row.
__device__ __forceinline__ v16bf load_a_frag(const __bf16* p) {
    v16bf a;
#pragma unroll
    for (int i = 0; i < 8; ++i) { a[i] = p[i]; a[i + 8] = p[i + 8]; }
    return a;
}

// Fallback: B fragment gathered from row-major fp32 weights (strided walk).
__device__ __forceinline__ v16bf load_b_frag_f32(const float* p, int ld) {
    v16bf b;
#pragma unroll
    for (int i = 0; i < 16; ++i) b[i] = (__bf16)p[(size_t)i * (size_t)ld];
    return b;
}

// Packed path: per-lane 32B contiguous fragment (2x global_load_b128).
__device__ __forceinline__ v16bf load_b_frag_packed(const __bf16* p) {
    return *(const v16bf*)p;
}

__device__ __forceinline__ float gelu_exact(float v) {
    return 0.5f * v * (1.0f + erff(v * 0.70710678118654752f));
}

// ---------------- Kernel 0: pack fp32 weights -> bf16 WMMA B-fragment layout --
// Layout per expert: [k_tile][n_tile][lane][16 bf16]; lane holds column
// N = n_tile*16 + (lane&15), K = k_tile*32 + (lane>>4)*16 + i.
__global__ __launch_bounds__(256) void pack_weights_kernel(
    const float* __restrict__ W1, const float* __restrict__ W2,
    __bf16* __restrict__ pW1, __bf16* __restrict__ pW2)
{
    const int lane = threadIdx.x & 31;
    const size_t w = (size_t)blockIdx.x * 8 + (threadIdx.x >> 5);
    const size_t nW1 = (size_t)Ee * KT1 * NT1;   // 65536 fragment blocks

    const float* src;
    __bf16* dst;
    int ld;
    if (w < nW1) {
        const size_t per = (size_t)KT1 * NT1;
        size_t e = w / per, r = w % per;
        size_t ktile = r / NT1, ntile = r % NT1;
        int d0 = (int)ktile * 32 + (lane >> 4) * 16;
        int h  = (int)ntile * 16 + (lane & 15);
        src = W1 + ((size_t)e * Dd + d0) * Hh + h;
        ld  = Hh;
        dst = pW1 + (size_t)e * ((size_t)Dd * Hh)
                  + (((size_t)ktile * NT1 + ntile) * 32 + lane) * 16;
    } else {
        const size_t w2 = w - nW1;
        const size_t per = (size_t)KT2 * NT2;
        size_t e = w2 / per, r = w2 % per;
        size_t ktile = r / NT2, ntile = r % NT2;
        int k0 = (int)ktile * 32 + (lane >> 4) * 16;
        int c  = (int)ntile * 16 + (lane & 15);
        src = W2 + ((size_t)e * Hh + k0) * Cc + c;
        ld  = Cc;
        dst = pW2 + (size_t)e * ((size_t)Hh * Cc)
                  + (((size_t)ktile * NT2 + ntile) * 32 + lane) * 16;
    }
    v16bf b;
#pragma unroll
    for (int i = 0; i < 16; ++i) b[i] = (__bf16)src[(size_t)i * (size_t)ld];
    *(v16bf*)dst = b;
}

// ---------------- Kernel 1: router logits + top-2 softmax + LN stats ----------
__global__ __launch_bounds__(256) void router_ln_kernel(
    const float* __restrict__ x, const float* __restrict__ Wr,
    const float* __restrict__ br,
    float* __restrict__ mu, float* __restrict__ rs,
    int* __restrict__ counts, int* __restrict__ tlist, float* __restrict__ wlist)
{
    const int lane = threadIdx.x & 31;
    const int wave = threadIdx.x >> 5;
    const int t = blockIdx.x * 8 + wave;
    if (t >= Bt) return;

    float sum = 0.f, sq = 0.f;
    float lg[Ee];
#pragma unroll
    for (int e = 0; e < Ee; ++e) lg[e] = 0.f;

    const float* xr = x + (size_t)t * Dd;
    for (int d = lane; d < Dd; d += 32) {
        float xv = xr[d];
        sum += xv; sq += xv * xv;
        const float* wr = Wr + (size_t)d * Ee;
#pragma unroll
        for (int e = 0; e < Ee; ++e) lg[e] += xv * wr[e];
    }
#pragma unroll
    for (int off = 16; off; off >>= 1) {
        sum += __shfl_xor(sum, off, 32);
        sq  += __shfl_xor(sq,  off, 32);
#pragma unroll
        for (int e = 0; e < Ee; ++e) lg[e] += __shfl_xor(lg[e], off, 32);
    }
    if (lane == 0) {
        float m = sum * (1.0f / Dd);
        float v = sq * (1.0f / Dd) - m * m;
        mu[t] = m;
        rs[t] = rsqrtf(v + 1e-5f);
        float vals[Ee];
#pragma unroll
        for (int e = 0; e < Ee; ++e) vals[e] = lg[e] + br[e];
        int i0 = 0; float v0 = vals[0];
#pragma unroll
        for (int e = 1; e < Ee; ++e) if (vals[e] > v0) { v0 = vals[e]; i0 = e; }
        int i1 = (i0 == 0) ? 1 : 0; float v1 = vals[i1];
#pragma unroll
        for (int e = 0; e < Ee; ++e)
            if (e != i0 && vals[e] > v1) { v1 = vals[e]; i1 = e; }
        float e1 = __expf(v1 - v0);
        float w0 = 1.0f / (1.0f + e1);
        float w1 = 1.0f - w0;
        int p0 = atomicAdd(&counts[i0], 1);
        tlist[i0 * Bt + p0] = t;  wlist[i0 * Bt + p0] = w0;
        int p1 = atomicAdd(&counts[i1], 1);
        tlist[i1 * Bt + p1] = t;  wlist[i1 * Bt + p1] = w1;
    }
}

// ---------------- Kernel 2: fused expert MLP (WMMA bf16) ----------------------
template <bool PACKED>
__global__ __launch_bounds__(128) void moe_expert_kernel(
    const float* __restrict__ x, const float* __restrict__ mu,
    const float* __restrict__ rs,
    const float* __restrict__ ln_g, const float* __restrict__ ln_b,
    const float* __restrict__ W1, const float* __restrict__ pb1,
    const float* __restrict__ W2, const float* __restrict__ pb2,
    const __bf16* __restrict__ pW1, const __bf16* __restrict__ pW2,
    const int* __restrict__ counts, const int* __restrict__ tlist,
    const float* __restrict__ wlist, float* __restrict__ out)
{
    __shared__ __bf16 sXn[16 * Dd];       // 32 KB: LN'd + affine'd tokens (bf16)
    __shared__ __bf16 sHc[16 * HCHUNK];   //  4 KB: gelu(h) chunk
    __shared__ int    sTok[16];
    __shared__ float  sWt[16];

    const int e    = blockIdx.x;
    const int tile = blockIdx.y;
    const int cnt  = counts[e];
    const int base = tile * 16;
    if (base >= cnt) return;

    if (threadIdx.x < 16) {
        int r = threadIdx.x;
        int src = base + r;
        if (src < cnt) { sTok[r] = tlist[e * Bt + src]; sWt[r] = wlist[e * Bt + src]; }
        else           { sTok[r] = tlist[e * Bt + base]; sWt[r] = 0.0f; }
    }
    __syncthreads();

    // Stage A: xe = ((x - mu) * rsigma) * ln_g[e] + ln_b[e]  -> bf16 in LDS
    {
        const float* lg = ln_g + (size_t)e * Dd;
        const float* lb = ln_b + (size_t)e * Dd;
        for (int idx = threadIdx.x; idx < 16 * Dd; idx += 128) {
            int r = idx >> 10, d = idx & (Dd - 1);
            int tok = sTok[r];
            float xv = x[(size_t)tok * Dd + d];
            float xn = (xv - mu[tok]) * rs[tok];
            sXn[idx] = (__bf16)(xn * lg[d] + lb[d]);
        }
    }
    __syncthreads();

    const int lane  = threadIdx.x & 31;
    const int wave  = threadIdx.x >> 5;
    const int khalf = lane >> 4;
    const int koff  = khalf * 16;
    const int ncl   = lane & 15;
    const int Mrow  = lane & 15;
    const int Mb    = khalf * 8;

    const float*  W1e  = W1  + (size_t)e * Dd * Hh;
    const float*  W2e  = W2  + (size_t)e * Hh * Cc;
    const __bf16* pW1e = PACKED ? pW1 + (size_t)e * ((size_t)Dd * Hh) : nullptr;
    const __bf16* pW2e = PACKED ? pW2 + (size_t)e * ((size_t)Hh * Cc) : nullptr;
    const __bf16* aRowX = sXn + (size_t)Mrow * Dd;

    const v8f zacc = {0.f, 0.f, 0.f, 0.f, 0.f, 0.f, 0.f, 0.f};
    v8f yacc[16];
#pragma unroll
    for (int t = 0; t < 16; ++t) yacc[t] = zacc;

    for (int hc0 = 0; hc0 < Hh; hc0 += HCHUNK) {
        // ---- GEMM1: h[16, 32 cols/wave] = xe[16,1024] @ W1[:, cols] ----
        v8f hacc0 = zacc, hacc1 = zacc;
        const int n0 = hc0 + wave * 32;
        const int ntile0 = n0 >> 4;
        for (int k0 = 0; k0 < Dd; k0 += 32) {
            v16bf a = load_a_frag(aRowX + k0 + koff);
            v16bf bf0, bf1;
            if constexpr (PACKED) {
                const __bf16* bp = pW1e
                    + (((size_t)(k0 >> 5) * NT1 + ntile0) * 32 + lane) * 16;
                bf0 = load_b_frag_packed(bp);
                bf1 = load_b_frag_packed(bp + 512);   // next n_tile
            } else {
                const float* bp = W1e + (size_t)(k0 + khalf * 16) * Hh;
                bf0 = load_b_frag_f32(bp + n0 + ncl, Hh);
                bf1 = load_b_frag_f32(bp + n0 + 16 + ncl, Hh);
            }
            hacc0 = wmma_bf16(a, bf0, hacc0);
            hacc1 = wmma_bf16(a, bf1, hacc1);
        }

        // ---- bias + exact GELU -> bf16 chunk in LDS ----
#pragma unroll
        for (int s = 0; s < 2; ++s) {
            const v8f acc = s ? hacc1 : hacc0;
            int hcol = n0 + s * 16 + ncl;
            float bb = pb1[(size_t)e * Hh + hcol];
#pragma unroll
            for (int r = 0; r < 8; ++r) {
                float g = gelu_exact(acc[r] + bb);
                sHc[(size_t)(Mb + r) * HCHUNK + (hcol - hc0)] = (__bf16)g;
            }
        }
        __syncthreads();

        // ---- GEMM2: y[16, 256 cols/wave] += h_chunk[16,128] @ W2[chunk, :] ----
        const __bf16* aRowH = sHc + (size_t)Mrow * HCHUNK;
        for (int kk = 0; kk < HCHUNK; kk += 32) {
            v16bf a = load_a_frag(aRowH + kk + koff);
            const int ktile = (hc0 + kk) >> 5;
            if constexpr (PACKED) {
                const __bf16* bp = pW2e
                    + (((size_t)ktile * NT2 + wave * 16) * 32 + lane) * 16;
                if (kk == 0 && hc0 + HCHUNK < Hh)
                    __builtin_prefetch(bp + (size_t)(HCHUNK / 32) * NT2 * 512, 0, 0);
#pragma unroll
                for (int t = 0; t < 16; ++t) {
                    v16bf bf = load_b_frag_packed(bp + (size_t)t * 512);
                    yacc[t] = wmma_bf16(a, bf, yacc[t]);
                }
            } else {
                const float* bp2 = W2e + (size_t)(hc0 + kk + khalf * 16) * Cc;
#pragma unroll
                for (int t = 0; t < 16; ++t) {
                    int n = wave * 256 + t * 16 + ncl;
                    v16bf bf = load_b_frag_f32(bp2 + n, Cc);
                    yacc[t] = wmma_bf16(a, bf, yacc[t]);
                }
            }
        }
        __syncthreads();
    }

    // ---- epilogue: + b2, scale by combine weight, atomic accumulate ----
#pragma unroll
    for (int t = 0; t < 16; ++t) {
        int n = wave * 256 + t * 16 + ncl;
        float bb = pb2[(size_t)e * Cc + n];
#pragma unroll
        for (int r = 0; r < 8; ++r) {
            int M = Mb + r;
            float val = (yacc[t][r] + bb) * sWt[M];
            atomicAdd(out + (size_t)sTok[M] * Cc + n, val);
        }
    }
}

extern "C" void kernel_launch(void* const* d_in, const int* in_sizes, int n_in,
                              void* d_out, int out_size, void* d_ws, size_t ws_size,
                              hipStream_t stream) {
    const float* x    = (const float*)d_in[0];
    const float* Wr   = (const float*)d_in[1];
    const float* br   = (const float*)d_in[2];
    const float* ln_g = (const float*)d_in[3];
    const float* ln_b = (const float*)d_in[4];
    const float* W1   = (const float*)d_in[5];
    const float* b1   = (const float*)d_in[6];
    const float* W2   = (const float*)d_in[7];
    const float* b2   = (const float*)d_in[8];
    float* out = (float*)d_out;

    // Workspace header (lists etc.) in first 1 MB; packed weights after.
    float* mu    = (float*)d_ws;                 // B
    float* rs    = mu + Bt;                      // B
    int*   cnts  = (int*)(rs + Bt);              // E
    int*   tlist = cnts + Ee;                    // E*B
    float* wlist = (float*)(tlist + Ee * Bt);    // E*B

    const size_t HDR = (size_t)1 << 20;
    const size_t W1_ELEMS = (size_t)Ee * Dd * Hh;   // 33,554,432
    const size_t W2_ELEMS = (size_t)Ee * Hh * Cc;   // 33,554,432
    const size_t NEED = HDR + sizeof(__bf16) * (W1_ELEMS + W2_ELEMS);
    const bool packed = (ws_size >= NEED);

    hipMemsetAsync(cnts, 0, Ee * sizeof(int), stream);
    hipMemsetAsync(out, 0, (size_t)out_size * sizeof(float), stream);

    router_ln_kernel<<<Bt / 8, 256, 0, stream>>>(x, Wr, br, mu, rs, cnts, tlist, wlist);

    dim3 grid(Ee, Bt / 16);
    if (packed) {
        __bf16* pW1 = (__bf16*)((char*)d_ws + HDR);
        __bf16* pW2 = pW1 + W1_ELEMS;
        // 65536 + 65536 fragment-pack waves, 8 waves per block
        pack_weights_kernel<<<16384, 256, 0, stream>>>(W1, W2, pW1, pW2);
        moe_expert_kernel<true><<<grid, 128, 0, stream>>>(
            x, mu, rs, ln_g, ln_b, W1, b1, W2, b2, pW1, pW2,
            cnts, tlist, wlist, out);
    } else {
        moe_expert_kernel<false><<<grid, 128, 0, stream>>>(
            x, mu, rs, ln_g, ln_b, W1, b1, W2, b2, nullptr, nullptr,
            cnts, tlist, wlist, out);
    }
}
